// GraphSAGEEncoder_6923487282696
// MI455X (gfx1250) — matrix operand order repaired and checked by reading, not compiled, not run on hardware
//
#include <hip/hip_runtime.h>
#include <hip/hip_bf16.h>
#include <stdint.h>

typedef __attribute__((ext_vector_type(16))) _Float16 v16h;
typedef __attribute__((ext_vector_type(8)))  _Float16 v8h;
typedef __attribute__((ext_vector_type(4)))  _Float16 v4h;
typedef __attribute__((ext_vector_type(8)))  float    v8f;
typedef __attribute__((ext_vector_type(4)))  float    v4f;

#define N_NODES  50000
#define N_EDGES  800000
#define CH       128
#define N_GRAPHS 512

// ---------------------------------------------------------------------------
// fp32 -> fp16 conversion (x and weight matrices)
// ---------------------------------------------------------------------------
__global__ void cvt_f32_to_f16_kernel(const float* __restrict__ src,
                                      _Float16* __restrict__ dst, int n) {
    int i = blockIdx.x * blockDim.x + threadIdx.x;
    if (i < n) dst[i] = (_Float16)src[i];
}

// ---------------------------------------------------------------------------
// in-degree via fp32 atomics
// ---------------------------------------------------------------------------
__global__ void degree_kernel(const int* __restrict__ edge, float* __restrict__ deg,
                              int n_edges) {
    int e = blockIdx.x * blockDim.x + threadIdx.x;
    if (e < n_edges) {
        int dst = edge[n_edges + e];
        atomicAdd(&deg[dst], 1.0f);
    }
}

__global__ void invdeg_kernel(float* __restrict__ deg, int n) {
    int i = blockIdx.x * blockDim.x + threadIdx.x;
    if (i < n) deg[i] = 1.0f / fmaxf(deg[i], 1.0f);
}

// ---------------------------------------------------------------------------
// scatter-add of neighbor features: agg[dst] += h16[src]
// 32 lanes per edge, 4 channels per lane (one 8-byte f16x4 load)
// ---------------------------------------------------------------------------
__global__ void scatter_kernel(const _Float16* __restrict__ h,
                               const int* __restrict__ edge,
                               float* __restrict__ agg, int n_edges) {
    int t = blockIdx.x * blockDim.x + threadIdx.x;
    int e = t >> 5;
    int lane = t & 31;
    if (e >= n_edges) return;
    int src = edge[e];
    int dst = edge[n_edges + e];
    int c = lane * 4;
    v4h v = *(const v4h*)(h + (size_t)src * CH + c);
    float* ap = agg + (size_t)dst * CH + c;
    atomicAdd(ap + 0, (float)v[0]);
    atomicAdd(ap + 1, (float)v[1]);
    atomicAdd(ap + 2, (float)v[2]);
    atomicAdd(ap + 3, (float)v[3]);
}

// ---------------------------------------------------------------------------
// Fused SAGE layer:  out = relu( (agg*inv_deg) @ Wl^T + h @ Wr^T + b )
// Tiling: block = 256 threads = 8 waves; blockIdx.x = M-tile (16 nodes);
// wave w = N-tile (16 output channels). K = 128 -> 4 k-tiles of 32.
// Per wave: 8 x v_wmma_f32_16x16x32_f16 into one f32 accumulator.
//
// 16-bit A 16x32 layout: lane L holds row M = L%16; K-half = L/16;
//   half idx 0..15 -> K = kb + (L/16)*16 + i (this code's contiguous order)
// 16-bit B 32x16 layout: lane L holds col N = L%16, K = kb + (L/16)*16 + i (this code's order, as A)
// f32 C/D 16x16: lane L holds col N = L%16, VGPR v -> row M = v + 8*(L/16)
// ---------------------------------------------------------------------------
__global__ void sage_wmma_kernel(const float*    __restrict__ agg,
                                 const float*    __restrict__ inv_deg,
                                 const _Float16* __restrict__ h,
                                 const _Float16* __restrict__ Wl,
                                 const _Float16* __restrict__ Wr,
                                 const float*    __restrict__ bias,
                                 _Float16*       __restrict__ out) {
    const int m0    = blockIdx.x * 16;
    const int wave  = threadIdx.x >> 5;
    const int lane  = threadIdx.x & 31;
    const int o0    = wave * 16;
    const int lm    = lane & 15;
    const int khalf = lane >> 4;

    const int m = m0 + lm;       // A row this lane supplies
    const int o = o0 + lm;       // B column (= output channel) this lane supplies
    const float inv  = inv_deg[m];
    const float bval = bias[o];  // C element depends only on N = lm

    v8f c;
#pragma unroll
    for (int i = 0; i < 8; i++) c[i] = bval;

#pragma unroll
    for (int kt = 0; kt < 4; kt++) {
        const int kb = kt * 32;

        // ---- A from agg (scale by inv_deg, convert to f16) ----
        const float* ar = agg + (size_t)m * CH + kb + khalf * 16;
        v4f f0 = *(const v4f*)(ar + 0);
        v4f f1 = *(const v4f*)(ar + 4);
        v4f f2 = *(const v4f*)(ar + 8);
        v4f f3 = *(const v4f*)(ar + 12);
        v16h a;
#pragma unroll
        for (int i = 0; i < 4; i++) {
            a[i]      = (_Float16)(f0[i] * inv);
            a[i + 4]  = (_Float16)(f1[i] * inv);
            a[i + 8]  = (_Float16)(f2[i] * inv);
            a[i + 12] = (_Float16)(f3[i] * inv);
        }
        v16h bl = *(const v16h*)(Wl + (size_t)o * CH + kb + khalf * 16);
        c = __builtin_amdgcn_wmma_f32_16x16x32_f16(false, a, false, bl,
                                                   (short)0, c, false, false);

        // ---- A from h (root/self term) ----
        const _Float16* hr = h + (size_t)m * CH + kb + khalf * 16;
        v8h lo = *(const v8h*)(hr);
        v8h hi = *(const v8h*)(hr + 8);
        v16h a2;
#pragma unroll
        for (int i = 0; i < 8; i++) { a2[i] = lo[i]; a2[i + 8] = hi[i]; }
        v16h br = *(const v16h*)(Wr + (size_t)o * CH + kb + khalf * 16);
        c = __builtin_amdgcn_wmma_f32_16x16x32_f16(false, a2, false, br,
                                                   (short)0, c, false, false);
    }

    // ---- ReLU + store f16 ----
#pragma unroll
    for (int v = 0; v < 8; v++) {
        int mrow = m0 + v + khalf * 8;
        float val = c[v] > 0.0f ? c[v] : 0.0f;
        out[(size_t)mrow * CH + o0 + lm] = (_Float16)val;
    }
}

// ---------------------------------------------------------------------------
// global mean pool: accumulate per-graph sums + counts
// ---------------------------------------------------------------------------
__global__ void pool_accum_kernel(const _Float16* __restrict__ h,
                                  const int* __restrict__ batch,
                                  float* __restrict__ pooled,
                                  float* __restrict__ counts, int n_nodes) {
    int t = blockIdx.x * blockDim.x + threadIdx.x;
    int node = t >> 5;
    int lane = t & 31;
    if (node >= n_nodes) return;
    int g = batch[node];
    int c = lane * 4;
    v4h v = *(const v4h*)(h + (size_t)node * CH + c);
    float* pp = pooled + (size_t)g * CH + c;
    atomicAdd(pp + 0, (float)v[0]);
    atomicAdd(pp + 1, (float)v[1]);
    atomicAdd(pp + 2, (float)v[2]);
    atomicAdd(pp + 3, (float)v[3]);
    if (lane == 0) atomicAdd(&counts[g], 1.0f);
}

__global__ void pool_final_kernel(const float* __restrict__ pooled,
                                  const float* __restrict__ counts,
                                  float* __restrict__ out) {
    int i = blockIdx.x * blockDim.x + threadIdx.x;
    if (i < N_GRAPHS * CH) {
        int g = i >> 7;  // /128
        out[i] = pooled[i] / fmaxf(counts[g], 1.0f);
    }
}

// ---------------------------------------------------------------------------
// launch
// ---------------------------------------------------------------------------
extern "C" void kernel_launch(void* const* d_in, const int* in_sizes, int n_in,
                              void* d_out, int out_size, void* d_ws, size_t ws_size,
                              hipStream_t stream) {
    const float* x     = (const float*)d_in[0];
    const int*   edge  = (const int*)d_in[1];   // [2, E] flat: [0..E)=src, [E..2E)=dst
    const int*   batch = (const int*)d_in[2];
    const float* Wf[6] = { (const float*)d_in[3], (const float*)d_in[4],   // W1l W1r
                           (const float*)d_in[6], (const float*)d_in[7],   // W2l W2r
                           (const float*)d_in[9], (const float*)d_in[10] };// W3l W3r
    const float* b1 = (const float*)d_in[5];
    const float* b2 = (const float*)d_in[8];
    const float* b3 = (const float*)d_in[11];

    // workspace layout (256B-aligned offsets)
    char* ws = (char*)d_ws;
    const size_t OFF_DEG   = 0;                                   // 50000 f32
    const size_t OFF_HA    = 200192;                              // 50000*128 f16
    const size_t OFF_HB    = OFF_HA + (size_t)N_NODES * CH * 2;   // 50000*128 f16
    const size_t OFF_W16   = OFF_HB + (size_t)N_NODES * CH * 2;   // 6*128*128 f16
    const size_t OFF_AGG   = OFF_W16 + 6 * 128 * 128 * 2;         // 50000*128 f32
    const size_t OFF_POOL  = OFF_AGG + (size_t)N_NODES * CH * 4;  // 512*128 f32
    const size_t OFF_CNT   = OFF_POOL + (size_t)N_GRAPHS * CH * 4;// 512 f32

    float*    deg    = (float*)(ws + OFF_DEG);
    _Float16* hA     = (_Float16*)(ws + OFF_HA);
    _Float16* hB     = (_Float16*)(ws + OFF_HB);
    _Float16* w16    = (_Float16*)(ws + OFF_W16);
    float*    agg    = (float*)(ws + OFF_AGG);
    float*    pooled = (float*)(ws + OFF_POOL);
    float*    counts = (float*)(ws + OFF_CNT);

    const int TB = 256;

    // degrees -> inverse degrees (in place)
    hipMemsetAsync(deg, 0, (size_t)N_NODES * 4, stream);
    degree_kernel<<<(N_EDGES + TB - 1) / TB, TB, 0, stream>>>(edge, deg, N_EDGES);
    invdeg_kernel<<<(N_NODES + TB - 1) / TB, TB, 0, stream>>>(deg, N_NODES);

    // fp16 conversions: x and 6 weight matrices
    cvt_f32_to_f16_kernel<<<((size_t)N_NODES * CH + TB - 1) / TB, TB, 0, stream>>>(
        x, hA, N_NODES * CH);
    for (int i = 0; i < 6; i++)
        cvt_f32_to_f16_kernel<<<(128 * 128 + TB - 1) / TB, TB, 0, stream>>>(
            Wf[i], w16 + (size_t)i * 128 * 128, 128 * 128);

    const float* biases[3] = { b1, b2, b3 };
    _Float16* hcur = hA;
    _Float16* hnxt = hB;
    const int scatter_blocks = ((size_t)N_EDGES * 32 + TB - 1) / TB;
    const int mtiles = N_NODES / 16;  // 3125 exact

    for (int layer = 0; layer < 3; layer++) {
        hipMemsetAsync(agg, 0, (size_t)N_NODES * CH * 4, stream);
        scatter_kernel<<<scatter_blocks, TB, 0, stream>>>(hcur, edge, agg, N_EDGES);
        sage_wmma_kernel<<<mtiles, TB, 0, stream>>>(
            agg, deg, hcur,
            w16 + (size_t)(2 * layer) * 128 * 128,      // Wl
            w16 + (size_t)(2 * layer + 1) * 128 * 128,  // Wr
            biases[layer], hnxt);
        _Float16* t = hcur; hcur = hnxt; hnxt = t;
    }

    // global mean pool
    hipMemsetAsync(pooled, 0, (size_t)N_GRAPHS * CH * 4 + N_GRAPHS * 4, stream);
    pool_accum_kernel<<<((size_t)N_NODES * 32 + TB - 1) / TB, TB, 0, stream>>>(
        hcur, batch, pooled, counts, N_NODES);
    pool_final_kernel<<<(N_GRAPHS * CH + TB - 1) / TB, TB, 0, stream>>>(
        pooled, counts, (float*)d_out);
}
